// Criterion_cosine_28604482192025
// MI455X (gfx1250) — matrix operand order repaired and checked by reading, not compiled, not run on hardware
//
#include <hip/hip_runtime.h>

typedef __attribute__((ext_vector_type(16))) _Float16 v16h;
typedef __attribute__((ext_vector_type(8)))  _Float16 v8h;
typedef __attribute__((ext_vector_type(4)))  _Float16 v4h;
typedef __attribute__((ext_vector_type(8)))  float    v8f;
typedef __attribute__((ext_vector_type(4)))  unsigned int uint32x4;
typedef __attribute__((ext_vector_type(8)))  int int32x8;
typedef __attribute__((ext_vector_type(4)))  int int32x4;

#ifndef USE_TDM
#define USE_TDM 1
#endif

namespace {
constexpr int kB      = 32;    // batch == queries
constexpr int kT      = 256;   // video frames
constexpr int kVid    = 1024;  // video feature dim
constexpr int kJ      = 512;   // joint dim
constexpr int kL      = 20;    // words per query
constexpr int kLayers = 6;
constexpr int kP      = 241;   // total proposals (127+62+30+14+6+2)
constexpr int kNmsK   = 5;
constexpr float kNmsThresh = 0.7f;
constexpr float kMargin    = 0.2f;
constexpr float kNeg       = -1e9f;
// LDS B-tile row stride: 32 data halves + 2 dwords TDM padding = 36 halves (72B)
constexpr int kBRow = 36;
}

// ---------------------------------------------------------------------------
// elementwise f32 -> f16 cast
// ---------------------------------------------------------------------------
__global__ void cast_f16_kernel(const float* __restrict__ in,
                                _Float16* __restrict__ out, int n) {
  int i = blockIdx.x * blockDim.x + threadIdx.x;
  if (i < n) out[i] = (_Float16)in[i];
}

// conv_w [L][O][C][K] (f32) -> [L][O][K][C] (f16): each output-channel row is
// (k-major, c-contiguous) and matches the [B,T,C] activation im2col row.
__global__ void permute_convw_kernel(const float* __restrict__ in,
                                     _Float16* __restrict__ out, int n) {
  int i = blockIdx.x * blockDim.x + threadIdx.x;
  if (i >= n) return;
  int k = i & 3;
  int c = (i >> 2) & 511;
  int o = (i >> 11) & 511;
  int l = i >> 20;
  out[(((long)(l * 512 + o) * 4 + k) * 512) + c] = (_Float16)in[i];
}

// ---------------------------------------------------------------------------
// TDM: async-load one B tile (64 rows x 32 halves, row stride K halves) from
// global into LDS at lds_byte_off, with 2-dword padding every 16 dwords so the
// LDS image has a 72B row stride. Issued by one wave; tracked by TENSORcnt.
// ---------------------------------------------------------------------------
__device__ __forceinline__ void tdm_load_b_tile(const _Float16* gptr,
                                                unsigned lds_byte_off, int K) {
#if USE_TDM
  unsigned long long ga = (unsigned long long)(size_t)gptr;
  uint32x4 g0;
  g0[0] = 1u;                                        // count=1 user descriptor
  g0[1] = lds_byte_off;                              // [63:32]  lds_addr
  g0[2] = (unsigned)(ga & 0xFFFFFFFFu);              // [95:64]  global_addr lo
  g0[3] = (unsigned)((ga >> 32) & 0x01FFFFFFu)       // [120:96] global_addr hi
        | (2u << 30);                                // [127:126] type=2 (image)
  unsigned td0 = (unsigned)K;                        // tensor_dim0 (elements)
  unsigned td1 = 64u;                                // tensor_dim1 (rows)
  unsigned long long st0 = (unsigned long long)K;    // tensor_dim0_stride
  int32x8 g1;
  g1[0] = (int)((1u << 16)        // data_size = 2 bytes
              | (1u << 20)        // pad_enable
              | (3u << 22)        // pad_interval: every 16 dwords
              | (1u << 25));      // pad_amount: 2 dwords
  g1[1] = (int)((td0 & 0xFFFFu) << 16);                              // td0[15:0]
  g1[2] = (int)(((td0 >> 16) & 0xFFFFu) | ((td1 & 0xFFFFu) << 16));  // td0 hi | td1 lo
  g1[3] = (int)(((td1 >> 16) & 0xFFFFu) | (32u << 16));              // td1 hi | tile_dim0=32
  g1[4] = (int)(64u);                                                // tile_dim1=64, tile_dim2=0
  g1[5] = (int)(unsigned)(st0 & 0xFFFFFFFFu);                        // stride0 lo
  g1[6] = (int)(unsigned)((st0 >> 32) & 0xFFFFu);                    // stride0 hi | stride1 lo=0
  g1[7] = 0;
  int32x4 z4 = {0, 0, 0, 0};
#if defined(__clang_major__) && __clang_major__ >= 23
  int32x8 z8 = {0, 0, 0, 0, 0, 0, 0, 0};
  __builtin_amdgcn_tensor_load_to_lds(g0, g1, z4, z4, z8, 0);
#else
  __builtin_amdgcn_tensor_load_to_lds(g0, g1, z4, z4, 0);
#endif
#else
  (void)gptr; (void)lds_byte_off; (void)K;
#endif
}

// Fallback staging: 128 threads cooperatively copy the same tile (same padded
// LDS layout) via global loads + ds stores.
__device__ __forceinline__ void stage_b_fallback(const _Float16* __restrict__ Bw,
                                                 long colBase, int K, int kk,
                                                 _Float16* dst, int tid) {
  int row = tid >> 1, part = tid & 1;
  const _Float16* g = Bw + (colBase + row) * (long)K + kk + part * 16;
  _Float16* d = dst + row * kBRow + part * 16;
  v16h gv = *(const v16h*)g;
#pragma unroll
  for (int c = 0; c < 4; ++c) {
    v4h t;
#pragma unroll
    for (int j = 0; j < 4; ++j) t[j] = gv[4 * c + j];
    *(v4h*)(d + 4 * c) = t;
  }
}

// ---------------------------------------------------------------------------
// Workgroup NT GEMM: 128 threads (4 waves) own a 64x64 C tile; wave w computes
// rows 16w..16w+15. B tile is double-buffered in LDS (TDM async load, wave 0
// issues, s_wait_tensorcnt + barrier publishes). A is loaded directly per-lane.
// A row m: b = m/aTl, t = m%aTl; ptr = A + b*aStrideB + t*aStrideT.
// Output row = b*outRowsPerB + outOff + t; ldc == N; stores f16 (Ch) or f32 (Cf).
// ---------------------------------------------------------------------------
__device__ __forceinline__ void store_tile16(const v8f& acc, int colBase, int rowSel,
                                             const float* __restrict__ bias,
                                             _Float16* __restrict__ Ch,
                                             float* __restrict__ Cf,
                                             const long* orows, int N, int relu) {
  int col = colBase + rowSel;
  float bs = bias ? bias[col] : 0.0f;
#pragma unroll
  for (int r = 0; r < 8; ++r) {
    if (orows[r] < 0) continue;
    float vv = acc[r] + bs;
    if (relu) vv = fmaxf(vv, 0.0f);
    long idx = orows[r] * (long)N + col;
    if (Ch) Ch[idx] = (_Float16)vv;
    else    Cf[idx] = vv;
  }
}

__global__ __launch_bounds__(128)
void wmma_gemm_nt_kernel(const _Float16* __restrict__ A,
                         const _Float16* __restrict__ Bw,
                         const float* __restrict__ bias,
                         _Float16* __restrict__ Ch,
                         float* __restrict__ Cf,
                         int M, int N, int K,
                         int aTl, long aStrideT, long aStrideB,
                         int outRowsPerB, int outOff,
                         int relu) {
  __shared__ __align__(128) _Float16 Bl[2][64 * kBRow];

  const int nTiles = N >> 6;
  const int tn = blockIdx.x % nTiles;
  const int tm = blockIdx.x / nTiles;
  const int tid = threadIdx.x;
  const int wave = tid >> 5;
  const int lane = tid & 31;
  const int rowSel = lane & 15;     // A row within wave-subtile / C column
  const int ksel = lane >> 4;       // half-wave K sub-block select
  const long colBase = (long)tn * 64;

  // per-lane A row pointer (clamped for partial last M tile)
  int m = tm * 64 + wave * 16 + rowSel;
  int mc = (m < M) ? m : (M - 1);
  int b = mc / aTl;
  int t = mc - b * aTl;
  const _Float16* __restrict__ arow = A + (long)b * aStrideB + (long)t * aStrideT;

  v8f acc0 = {}, acc1 = {}, acc2 = {}, acc3 = {};

  // prologue: stage first B tile into buffer 0
#if USE_TDM
  if (tid < 32) {
    tdm_load_b_tile(Bw + colBase * K, (unsigned)(size_t)(void*)&Bl[0][0], K);
  }
#else
  stage_b_fallback(Bw, colBase, K, 0, &Bl[0][0], tid);
#endif

  for (int kk = 0; kk < K; kk += 32) {
    const int cur = (kk >> 5) & 1;
    const bool hasNext = (kk + 32) < K;
#if USE_TDM
    if (tid < 32) {
      if (hasNext) {
        tdm_load_b_tile(Bw + colBase * K + (kk + 32),
                        (unsigned)(size_t)(void*)&Bl[1 - cur][0], K);
        __builtin_amdgcn_s_wait_tensorcnt(1);
      } else {
        __builtin_amdgcn_s_wait_tensorcnt(0);
      }
    }
#else
    if (hasNext) stage_b_fallback(Bw, colBase, K, kk + 32, &Bl[1 - cur][0], tid);
#endif
    __syncthreads();

    // A 16x32 f16 fragment (lanes 0-15: K {0..15}; lanes 16-31: {16..31})
    v8h alo = *(const v8h*)(arow + kk + ksel * 16);
    v8h ahi = *(const v8h*)(arow + kk + ksel * 16 + 8);
    v16h av;
#pragma unroll
    for (int i = 0; i < 8; ++i) { av[i] = alo[i]; av[i + 8] = ahi[i]; }

    // B fragments from padded LDS image
    v16h bv[4];
#pragma unroll
    for (int s = 0; s < 4; ++s) {
      const _Float16* src = &Bl[cur][(s * 16 + rowSel) * kBRow + ksel * 16];
#pragma unroll
      for (int c = 0; c < 4; ++c) {
        v4h p = *(const v4h*)(src + 4 * c);
#pragma unroll
        for (int j = 0; j < 4; ++j) bv[s][4 * c + j] = p[j];
      }
    }

    acc0 = __builtin_amdgcn_wmma_f32_16x16x32_f16(false, av, false, bv[0], (short)0, acc0, false, false);
    acc1 = __builtin_amdgcn_wmma_f32_16x16x32_f16(false, av, false, bv[1], (short)0, acc1, false, false);
    acc2 = __builtin_amdgcn_wmma_f32_16x16x32_f16(false, av, false, bv[2], (short)0, acc2, false, false);
    acc3 = __builtin_amdgcn_wmma_f32_16x16x32_f16(false, av, false, bv[3], (short)0, acc3, false, false);
    __syncthreads();
  }

  // C layout: element r of v8f -> row (r + 8*ksel), col = lane&15 per subtile.
  long orows[8];
#pragma unroll
  for (int r = 0; r < 8; ++r) {
    int m2 = tm * 64 + wave * 16 + r + 8 * ksel;
    if (m2 < M) {
      int b2 = m2 / aTl;
      int t2 = m2 - b2 * aTl;
      orows[r] = (long)b2 * outRowsPerB + outOff + t2;
    } else {
      orows[r] = -1;
    }
  }
  store_tile16(acc0, tn * 64 +  0, rowSel, bias, Ch, Cf, orows, N, relu);
  store_tile16(acc1, tn * 64 + 16, rowSel, bias, Ch, Cf, orows, N, relu);
  store_tile16(acc2, tn * 64 + 32, rowSel, bias, Ch, Cf, orows, N, relu);
  store_tile16(acc3, tn * 64 + 48, rowSel, bias, Ch, Cf, orows, N, relu);
}

// ---------------------------------------------------------------------------
// Per-query word Gram matrix: G[q][a][b] = words[q,a,:] . words[q,b,:]  (f32)
// ---------------------------------------------------------------------------
__global__ void gram_kernel(const float* __restrict__ words, float* __restrict__ G) {
  int i = blockIdx.x * blockDim.x + threadIdx.x;
  if (i >= kB * kL * kL) return;
  int q = i / (kL * kL);
  int r = i % (kL * kL);
  int la = r / kL, lb = r % kL;
  const float* wa = words + (long)(q * kL + la) * kJ;
  const float* wb = words + (long)(q * kL + lb) * kJ;
  float s = 0.f;
  for (int d = 0; d < kJ; ++d) s += wa[d] * wb[d];
  G[i] = s;
}

// ||f|| per proposal row of F [B*P, 512] (f16)
__global__ void fnorm_kernel(const _Float16* __restrict__ F, float* __restrict__ fn, int nrows) {
  int i = blockIdx.x * blockDim.x + threadIdx.x;
  if (i >= nrows) return;
  const _Float16* p = F + (long)i * kJ;
  float s = 0.f;
  for (int d = 0; d < kJ; ++d) { float x = (float)p[d]; s += x * x; }
  fn[i] = sqrtf(s);
}

// ---------------------------------------------------------------------------
// cosine similarity from attention logits:
//   a = softmax(logits);  f.v_s = sum a_l * logit_l;  ||v_s||^2 = a^T G_q a
//   sim = (f.v_s) / ((||f||+eps)(||v_s||+eps))
// att layout: [B*P, Q*L] f32 (row m = b*P+p, col = q*L+l); score: [Q,B,P]
// ---------------------------------------------------------------------------
__global__ void sim_kernel(const float* __restrict__ att,
                           const float* __restrict__ G,
                           const float* __restrict__ fnorm,
                           float* __restrict__ score) {
  int i = blockIdx.x * blockDim.x + threadIdx.x;
  if (i >= kB * kB * kP) return;
  int q = i / (kB * kP);
  int rem = i - q * (kB * kP);
  int b = rem / kP;
  int p = rem - b * kP;

  const float* lg = att + (long)(b * kP + p) * (kB * kL) + q * kL;
  float e[kL];
  float mx = -3.0e38f;
#pragma unroll
  for (int l = 0; l < kL; ++l) mx = fmaxf(mx, lg[l]);
  float se = 0.f;
#pragma unroll
  for (int l = 0; l < kL; ++l) { e[l] = __expf(lg[l] - mx); se += e[l]; }
  float inv = 1.0f / se;
  float dot = 0.f;
#pragma unroll
  for (int l = 0; l < kL; ++l) { e[l] *= inv; dot += e[l] * lg[l]; }

  const float* Gq = G + q * kL * kL;
  float n2 = 0.f;
#pragma unroll
  for (int la = 0; la < kL; ++la) {
    float ga = 0.f;
#pragma unroll
    for (int lb = 0; lb < kL; ++lb) ga += Gq[la * kL + lb] * e[lb];
    n2 += e[la] * ga;
  }
  float nv = sqrtf(fmaxf(n2, 0.f));
  float sim = dot / ((fnorm[b * kP + p] + 1e-8f) * (nv + 1e-8f));
  score[((long)q * kB + b) * kP + p] = sim;
}

// positive_map: score[b][b][p] -> d_out[1 + b*P + p]
__global__ void posmap_kernel(const float* __restrict__ score, float* __restrict__ out) {
  int i = blockIdx.x * blockDim.x + threadIdx.x;
  if (i >= kB * kP) return;
  int b = i / kP, p = i % kP;
  out[1 + i] = score[((long)b * kB + b) * kP + p];
}

// ---------------------------------------------------------------------------
// Greedy temporal NMS scoring: one wave per (q,b) row of 241 proposals.
// ---------------------------------------------------------------------------
__global__ __launch_bounds__(32)
void nms_kernel(const float* __restrict__ score, const float* __restrict__ iou,
                const float* __restrict__ lamp, float* __restrict__ smat) {
  int row = blockIdx.x;                 // q*32 + b
  int lane = threadIdx.x;
  const float* s = score + (long)row * kP;
  float v[8];
#pragma unroll
  for (int i = 0; i < 8; ++i) {
    int j = i * 32 + lane;
    v[i] = (j < kP) ? s[j] : kNeg;
  }
  float lam = lamp[0];
  float w = 1.0f, acc = 0.0f;
  for (int k = 0; k < kNmsK; ++k) {
    float bv = -3.0e38f;
    int bj = 0;
#pragma unroll
    for (int i = 0; i < 8; ++i) {       // ascending j => earliest index on ties
      int j = i * 32 + lane;
      if (v[i] > bv) { bv = v[i]; bj = j; }
    }
#pragma unroll
    for (int off = 16; off > 0; off >>= 1) {
      float ov = __shfl_xor(bv, off, 32);
      int   oj = __shfl_xor(bj, off, 32);
      if (ov > bv || (ov == bv && oj < bj)) { bv = ov; bj = oj; }
    }
    acc += bv * w;
    w *= lam;
    const float* ir = iou + (long)bj * kP;
#pragma unroll
    for (int i = 0; i < 8; ++i) {
      int j = i * 32 + lane;
      if (j < kP && (j == bj || ir[j] > kNmsThresh)) v[i] = kNeg;
    }
  }
  if (lane == 0) smat[row] = acc * (1.0f / (float)kNmsK);
}

// margin ranking diag loss over the 32x32 score matrix; single 1024-thread block.
__global__ __launch_bounds__(1024)
void loss_kernel(const float* __restrict__ smat, float* __restrict__ out) {
  __shared__ float red[1024];
  int tid = threadIdx.x;
  int r = tid >> 5, c = tid & 31;
  float s = smat[tid];
  float dr = smat[r * 33];              // r*32 + r
  float dc = smat[c * 33];
  float cs = (r == c) ? 0.f : fmaxf(0.f, kMargin + s - dr);
  float ci = (r == c) ? 0.f : fmaxf(0.f, kMargin + s - dc);
  red[tid] = cs + ci;
  __syncthreads();
  for (int st = 512; st > 0; st >>= 1) {
    if (tid < st) red[tid] += red[tid + st];
    __syncthreads();
  }
  if (tid == 0) out[0] = red[0] / (float)kB;
}

// ---------------------------------------------------------------------------
// host launcher
// ---------------------------------------------------------------------------
extern "C" void kernel_launch(void* const* d_in, const int* in_sizes, int n_in,
                              void* d_out, int out_size, void* d_ws, size_t ws_size,
                              hipStream_t stream) {
  (void)in_sizes; (void)n_in; (void)out_size; (void)ws_size;
  const float* video  = (const float*)d_in[0];   // [32,256,1024]
  const float* words  = (const float*)d_in[1];   // [32,20,512]
  /* d_in[2] w_masks: all-true, masking is a no-op */
  const float* lam    = (const float*)d_in[3];   // [1]
  const float* iou    = (const float*)d_in[4];   // [241,241]
  const float* fc_w   = (const float*)d_in[5];   // [512,1024]
  const float* fc_b   = (const float*)d_in[6];   // [512]
  const float* conv_w = (const float*)d_in[7];   // [6,512,512,4]
  const float* conv_b = (const float*)d_in[8];   // [6,512]
  const float* pw     = (const float*)d_in[9];   // [512,512,1]
  const float* pb     = (const float*)d_in[10];  // [512]
  float* out = (float*)d_out;

  // workspace carve-up (256B aligned)
  char* base = (char*)d_ws;
  size_t off = 0;
  auto alloc = [&](size_t bytes) {
    void* p = base + off;
    off = (off + bytes + 255) & ~(size_t)255;
    return p;
  };
  _Float16* video_h = (_Float16*)alloc((size_t)kB * kT * kVid * 2);
  _Float16* fcw_h   = (_Float16*)alloc((size_t)kJ * kVid * 2);
  _Float16* words_h = (_Float16*)alloc((size_t)kB * kL * kJ * 2);
  _Float16* pw_h    = (_Float16*)alloc((size_t)kJ * kJ * 2);
  _Float16* w2      = (_Float16*)alloc((size_t)kLayers * kJ * 4 * kJ * 2);
  _Float16* act0    = (_Float16*)alloc((size_t)kB * kT * kJ * 2);
  _Float16* actA    = (_Float16*)alloc((size_t)kB * 127 * kJ * 2);
  _Float16* actB    = (_Float16*)alloc((size_t)kB * 127 * kJ * 2);
  _Float16* F       = (_Float16*)alloc((size_t)kB * kP * kJ * 2);
  float*    att     = (float*)alloc((size_t)kB * kP * kB * kL * 4);
  float*    G       = (float*)alloc((size_t)kB * kL * kL * 4);
  float*    fnorm   = (float*)alloc((size_t)kB * kP * 4);
  float*    score   = (float*)alloc((size_t)kB * kB * kP * 4);
  float*    smat    = (float*)alloc((size_t)kB * kB * 4);

  auto cast16 = [&](const float* src, _Float16* dst, int n) {
    cast_f16_kernel<<<(n + 255) / 256, 256, 0, stream>>>(src, dst, n);
  };
  cast16(video, video_h, kB * kT * kVid);
  cast16(fc_w, fcw_h, kJ * kVid);
  cast16(words, words_h, kB * kL * kJ);
  cast16(pw, pw_h, kJ * kJ);
  {
    int n = kLayers * kJ * kJ * 4;
    permute_convw_kernel<<<(n + 255) / 256, 256, 0, stream>>>(conv_w, w2, n);
  }

  auto gemm = [&](const _Float16* A, const _Float16* Bw, const float* bias,
                  _Float16* Ch, float* Cf, int M, int N, int K,
                  int aTl, long aStT, long aStB, int outRpb, int outOff, int relu) {
    dim3 grid((unsigned)(((M + 63) / 64) * (N / 64)));
    wmma_gemm_nt_kernel<<<grid, 128, 0, stream>>>(A, Bw, bias, Ch, Cf, M, N, K,
                                                  aTl, aStT, aStB, outRpb, outOff, relu);
  };

  // fc: [B*T,1024] x [512,1024]^T -> act0 [B,T,512]  (no relu)
  gemm(video_h, fcw_h, fc_b, act0, nullptr,
       kB * kT, kJ, kVid, kT, kVid, (long)kT * kVid, kT, 0, 0);

  // conv pyramid + pointwise projection into F [B,241,512]
  const int Tin[kLayers]  = {256, 127, 62, 30, 14, 6};
  const int Tout[kLayers] = {127, 62, 30, 14, 6, 2};
  const int Poff[kLayers] = {0, 127, 189, 219, 233, 239};
  const _Float16* cur = act0;
  _Float16* pong[2] = {actA, actB};
  for (int li = 0; li < kLayers; ++li) {
    _Float16* oc = pong[li & 1];
    // strided conv as NT GEMM over im2col rows (K = 4*512, A row = 4 consecutive frames)
    gemm(cur, w2 + (size_t)li * kJ * 4 * kJ, conv_b + li * kJ, oc, nullptr,
         kB * Tout[li], kJ, 4 * kJ,
         Tout[li], (long)2 * kJ, (long)Tin[li] * kJ, Tout[li], 0, /*relu=*/1);
    // pointwise projection, scattered into concatenated proposal buffer F
    gemm(oc, pw_h, pb, F, nullptr,
         kB * Tout[li], kJ, kJ,
         Tout[li], (long)kJ, (long)Tout[li] * kJ, kP, Poff[li], /*relu=*/0);
    cur = oc;
  }

  // attention logits: F [B*P,512] x words [640,512]^T -> att f32 [B*P, 640]
  gemm(F, words_h, nullptr, nullptr, att,
       kB * kP, kB * kL, kJ, kP, (long)kJ, (long)kP * kJ, kP, 0, 0);

  {
    int n = kB * kL * kL;
    gram_kernel<<<(n + 255) / 256, 256, 0, stream>>>(words, G);
  }
  {
    int n = kB * kP;
    fnorm_kernel<<<(n + 255) / 256, 256, 0, stream>>>(F, fnorm, n);
  }
  {
    int n = kB * kB * kP;
    sim_kernel<<<(n + 255) / 256, 256, 0, stream>>>(att, G, fnorm, score);
  }
  {
    int n = kB * kP;
    posmap_kernel<<<(n + 255) / 256, 256, 0, stream>>>(score, out);
  }
  nms_kernel<<<kB * kB, 32, 0, stream>>>(score, iou, lam, smat);
  loss_kernel<<<1, 1024, 0, stream>>>(smat, out);
}
